// PointNetfeat2_78658031059298
// MI455X (gfx1250) — matrix operand order repaired and checked by reading, not compiled, not run on hardware
//
#include <hip/hip_runtime.h>
#include <hip/hip_bf16.h>

typedef _Float16 h16;
typedef __attribute__((ext_vector_type(16))) _Float16 v16h;
typedef __attribute__((ext_vector_type(8)))  _Float16 v8h;
typedef __attribute__((ext_vector_type(8)))  float    v8f;

// builtin parameter types per hipcc diagnostic: pointer to int4 (vector_size style)
typedef int v4i_ __attribute__((vector_size(16)));
typedef __attribute__((address_space(1))) v4i_ gbl_v4i;
typedef __attribute__((address_space(3))) v4i_ lds_v4i;

#define BATCH 16
#define NPTS  8192

#if __has_builtin(__builtin_amdgcn_global_load_async_to_lds_b128)
#define HAVE_ASYNC_LDS 1
#else
#define HAVE_ASYNC_LDS 0
#endif

#if HAVE_ASYNC_LDS
#if __has_builtin(__builtin_amdgcn_s_wait_asynccnt)
#define WAIT_ASYNC(n) __builtin_amdgcn_s_wait_asynccnt(n)
#else
#define WAIT_ASYNC(n) asm volatile("s_wait_asynccnt " #n ::: "memory")
#endif
#else
#define WAIT_ASYNC(n)
#endif

// async (or fallback sync) 16-byte global -> LDS copy
__device__ __forceinline__ void copy16_to_lds(const h16* g, h16* l) {
#if HAVE_ASYNC_LDS
  __builtin_amdgcn_global_load_async_to_lds_b128(
      (gbl_v4i*)(uintptr_t)g,
      (lds_v4i*)(uint32_t)(uintptr_t)l, 0, 0);
#else
  *(v8h*)l = *(const v8h*)g;
#endif
}

// ---------------- layout transform: x [B,3,N] -> xyz [B,N,3] ----------------
__global__ void k_transpose(const float* __restrict__ x, float* __restrict__ xyz, int n) {
  int i = blockIdx.x * blockDim.x + threadIdx.x;           // over B*N
  if (i >= BATCH * n) return;
  int b = i / n, p = i - b * n;
  const float* src = x + (size_t)b * 3 * n;
  float* dst = xyz + (size_t)i * 3;
  dst[0] = src[p];
  dst[1] = src[n + p];
  dst[2] = src[2 * n + p];
}

// ---------------- farthest point sampling: one block per batch ----------------
__global__ __launch_bounds__(256) void k_fps(const float* __restrict__ xyz,
                                             int* __restrict__ fidx, int n, int s) {
  __shared__ float dist[NPTS];
  __shared__ float rv[256];
  __shared__ int   ri[256];
  int b = blockIdx.x, t = threadIdx.x;
  const float* p = xyz + (size_t)b * n * 3;
  for (int i = t; i < n; i += 256) dist[i] = 1e10f;
  __syncthreads();
  int far = 0;
  for (int it = 0; it < s; ++it) {
    if (t == 0) fidx[b * s + it] = far;
    float cx = p[far * 3 + 0], cy = p[far * 3 + 1], cz = p[far * 3 + 2];
    float bm = -1.f; int bi = 0;
    for (int i = t; i < n; i += 256) {
      float dx = p[i * 3 + 0] - cx, dy = p[i * 3 + 1] - cy, dz = p[i * 3 + 2] - cz;
      float d = dx * dx + dy * dy + dz * dz;
      float nd = fminf(dist[i], d);
      dist[i] = nd;
      if (nd > bm) { bm = nd; bi = i; }
    }
    rv[t] = bm; ri[t] = bi;
    __syncthreads();
    for (int off = 128; off > 0; off >>= 1) {
      if (t < off && rv[t + off] > rv[t]) { rv[t] = rv[t + off]; ri[t] = ri[t + off]; }
      __syncthreads();
    }
    far = ri[0];
    __syncthreads();
  }
}

// ---------------- gather sampled centroids ----------------
__global__ void k_gather_xyz(const float* __restrict__ xyz, const int* __restrict__ fidx,
                             float* __restrict__ nxyz, int n, int s) {
  int i = blockIdx.x * blockDim.x + threadIdx.x;           // over B*s
  if (i >= BATCH * s) return;
  int b = i / s;
  int g = fidx[i];
  const float* src = xyz + ((size_t)b * n + g) * 3;
  nxyz[i * 3 + 0] = src[0];
  nxyz[i * 3 + 1] = src[1];
  nxyz[i * 3 + 2] = src[2];
}

// ---------------- ball query (torch semantics: K smallest in-range indices) ----------------
__global__ void k_query_ball(const float* __restrict__ xyz, const float* __restrict__ nxyz,
                             int* __restrict__ gi, int nq, int s, int n, int K, float r2) {
  int q = blockIdx.x * blockDim.x + threadIdx.x;
  if (q >= nq) return;
  int b = q / s;
  const float* p = xyz + (size_t)b * n * 3;
  float cx = nxyz[q * 3 + 0], cy = nxyz[q * 3 + 1], cz = nxyz[q * 3 + 2];
  int* out = gi + (size_t)q * K;
  int cnt = 0;
  for (int i = 0; i < n && cnt < K; ++i) {
    float dx = p[i * 3 + 0] - cx, dy = p[i * 3 + 1] - cy, dz = p[i * 3 + 2] - cz;
    if (dx * dx + dy * dy + dz * dz <= r2) out[cnt++] = i;
  }
  int f = (cnt > 0) ? out[0] : 0;   // FPS centroid guarantees cnt>=1
  for (; cnt < K; ++cnt) out[cnt] = f;
}

// ---- fold BN into weights, store TRANSPOSED f16 Wt[Cout][Kpad]; bias tb f32 ----
__global__ void k_prep_w(const float* __restrict__ W, const float* __restrict__ bb,
                         const float* __restrict__ g, const float* __restrict__ bt,
                         h16* __restrict__ Wt, float* __restrict__ tb,
                         int kin, int kpad, int cout) {
  const float inv = 0.99999500003749969f;  // 1/sqrt(1+1e-5)
  int i = blockIdx.x * blockDim.x + threadIdx.x;
  if (i < cout) tb[i] = bb[i] * inv * g[i] + bt[i];
  int tot = kpad * cout;
  if (i >= tot) return;
  int k = i / cout, c = i - k * cout;
  float w = (k < kin) ? W[k * cout + c] * inv * g[c] : 0.f;
  Wt[(size_t)c * kpad + k] = (h16)w;
}

// ---------------- grouping kernels: build padded f16 activation rows ----------------
__global__ void k_group_sa1(const float* __restrict__ xyz, const float* __restrict__ nxyz,
                            const int* __restrict__ gi, h16* __restrict__ A,
                            int row0, int rows, int K, int S, int n) {
  int m = blockIdx.x * blockDim.x + threadIdx.x;
  if (m >= rows * K) return;
  int rs = m / K, k = m - rs * K;
  int q = row0 + rs;
  int b = q / S;
  int g = gi[(size_t)q * K + k];
  const float* p = xyz + ((size_t)b * n + g) * 3;
  h16* a = A + (size_t)m * 32;
  a[0] = (h16)(p[0] - nxyz[q * 3 + 0]);
  a[1] = (h16)(p[1] - nxyz[q * 3 + 1]);
  a[2] = (h16)(p[2] - nxyz[q * 3 + 2]);
#pragma unroll
  for (int c = 3; c < 32; ++c) a[c] = (h16)0.f;
}

__global__ void k_group_sa2(const float* __restrict__ xyz, const float* __restrict__ nxyz,
                            const h16* __restrict__ pts, const int* __restrict__ gi,
                            h16* __restrict__ A,
                            int row0, int rows, int K, int S, int n, int D, int Kp) {
  int m = blockIdx.x * blockDim.x + threadIdx.x;
  if (m >= rows * K) return;
  int rs = m / K, k = m - rs * K;
  int q = row0 + rs;
  int b = q / S;
  int g = gi[(size_t)q * K + k];
  const h16* f = pts + ((size_t)b * n + g) * D;
  h16* a = A + (size_t)m * Kp;
  // D=320 halfs, both rows 16B aligned -> 40 x 16B vector copies
  for (int c = 0; c < D / 8; ++c) ((v8h*)a)[c] = ((const v8h*)f)[c];
  const float* p = xyz + ((size_t)b * n + g) * 3;
  a[D + 0] = (h16)(p[0] - nxyz[q * 3 + 0]);
  a[D + 1] = (h16)(p[1] - nxyz[q * 3 + 1]);
  a[D + 2] = (h16)(p[2] - nxyz[q * 3 + 2]);
  for (int c = D + 3; c < Kp; ++c) a[c] = (h16)0.f;
}

__global__ void k_group_sa3(const float* __restrict__ nxyz2, const h16* __restrict__ l2,
                            h16* __restrict__ A, int M, int D, int Kp) {
  int m = blockIdx.x * blockDim.x + threadIdx.x;    // over B*128
  if (m >= M) return;
  h16* a = A + (size_t)m * Kp;
  a[0] = (h16)nxyz2[m * 3 + 0];
  a[1] = (h16)nxyz2[m * 3 + 1];
  a[2] = (h16)nxyz2[m * 3 + 2];
  const h16* f = l2 + (size_t)m * D;
  for (int c = 0; c < D; ++c) a[3 + c] = f[c];
  for (int c = 3 + D; c < Kp; ++c) a[c] = (h16)0.f;
}

// ---------------- WMMA GEMM + bias + ReLU ----------------
// Out[M,Cout] = relu(A[M,Kp] x W[Kp,Cout] + tb), W given transposed as Wt[Cout][Kp].
// Block = 256 thr = 8 waves; tile 128(M) x 32(N); wave = 16x32 (two 16x16x32 WMMA per K-step).
// A tile double-buffered in LDS via async global->LDS (ASYNCcnt); B fragments read
// directly from global (weights are L2-resident, each lane's fragment is 32B contiguous).
// M%128==0, Kp%32==0, Cout%32==0 by construction -> no divergence, EXEC all ones.
__global__ __launch_bounds__(256) void k_gemm_relu(const h16* __restrict__ A,
                                                   const h16* __restrict__ Wt,
                                                   const float* __restrict__ tb,
                                                   h16* __restrict__ Out,
                                                   int Kp, int Cout) {
  __shared__ h16 As[2][128 * 32];   // 2 x 8KB A tiles, row-major [r][k]
  int tid  = threadIdx.x;
  int lane = tid & 31, wv = tid >> 5;
  int m0 = blockIdx.x * 128;
  int n0 = blockIdx.y * 32;
  int half = lane >> 4;             // 0: lanes 0-15, 1: lanes 16-31
  int mr   = lane & 15;
  int kb    = half * 16;            // A fragment K-base (column-contiguous K)
  int kbase = half * 16;            // B fragment K-base (column-contiguous K)

  // per-thread A-copy chunks: c = tid and tid+256; row = c>>2, half-off = (c&3)*8
  int r0 = tid >> 2;
  int r1 = (tid + 256) >> 2;
  int oo = (tid & 3) * 8;
  const h16* g0 = A + (size_t)(m0 + r0) * Kp + oo;
  const h16* g1 = A + (size_t)(m0 + r1) * Kp + oo;
  h16* l0a = &As[0][r0 * 32 + oo];
  h16* l1a = &As[0][r1 * 32 + oo];
  h16* l0b = &As[1][r0 * 32 + oo];
  h16* l1b = &As[1][r1 * 32 + oo];

  const h16* b0p = Wt + (size_t)(n0 + mr) * Kp + kbase;
  const h16* b1p = Wt + (size_t)(n0 + 16 + mr) * Kp + kbase;
  int arow = (wv * 16 + mr) * 32;

  v8f c0 = {}; v8f c1 = {};
  int nk = Kp >> 5;

  // prologue: stage 0 into buffer 0
  copy16_to_lds(g0, l0a);
  copy16_to_lds(g1, l1a);

  for (int i = 0; i < nk; ++i) {
    int buf = i & 1;
    if (i + 1 < nk) {           // prefetch next K-step into the other buffer
      const h16* ng0 = g0 + (size_t)(i + 1) * 32;
      const h16* ng1 = g1 + (size_t)(i + 1) * 32;
      copy16_to_lds(ng0, buf ? l0a : l0b);
      copy16_to_lds(ng1, buf ? l1a : l1b);
      WAIT_ASYNC(2);            // stage i complete; stage i+1 may be in flight
    } else {
      WAIT_ASYNC(0);
    }
    __syncthreads();            // all waves' stage-i LDS writes visible

    // B fragments straight from global: 32B contiguous per lane
    v16h bv0 = *(const v16h*)(b0p + (size_t)i * 32);
    v16h bv1 = *(const v16h*)(b1p + (size_t)i * 32);
    // A fragment from LDS: lanes 0-15 K=[0..16), lanes 16-31 K=[16..32)
    v8h alo = *(const v8h*)(&As[buf][arow + kb]);
    v8h ahi = *(const v8h*)(&As[buf][arow + kb + 8]);
    v16h a;
#pragma unroll
    for (int e = 0; e < 8; ++e) { a[e] = alo[e]; a[8 + e] = ahi[e]; }

    c0 = __builtin_amdgcn_wmma_f32_16x16x32_f16(false, a, false, bv0, (short)0, c0, false, false);
    c1 = __builtin_amdgcn_wmma_f32_16x16x32_f16(false, a, false, bv1, (short)0, c1, false, false);
    __syncthreads();            // reads done before buffer is overwritten next iter
  }

  // epilogue: C/D layout -> lane n = lane&15, VGPR r -> m = r + 8*half
  int n = lane & 15;
  float t0 = tb[n0 + n];
  float t1 = tb[n0 + 16 + n];
#pragma unroll
  for (int r = 0; r < 8; ++r) {
    int m = m0 + wv * 16 + half * 8 + r;
    float y0 = fmaxf(c0[r] + t0, 0.f);
    float y1 = fmaxf(c1[r] + t1, 0.f);
    Out[(size_t)m * Cout + n0 + n]      = (h16)y0;
    Out[(size_t)m * Cout + n0 + 16 + n] = (h16)y1;
  }
}

// ---------------- max over K neighbors, write into concat feature buffer ----------------
__global__ void k_maxpool_cat(const h16* __restrict__ feat, h16* __restrict__ out,
                              int row0, int rows, int K, int C, int Dtot, int coloff) {
  int i = blockIdx.x * blockDim.x + threadIdx.x;
  if (i >= rows * C) return;
  int rs = i / C, d = i - rs * C;
  const h16* f = feat + ((size_t)rs * K) * C + d;
  float mx = -3.0e38f;
  for (int k = 0; k < K; ++k) mx = fmaxf(mx, (float)f[(size_t)k * C]);
  out[(size_t)(row0 + rs) * Dtot + coloff + d] = (h16)mx;
}

__global__ void k_maxpool_final(const h16* __restrict__ feat, float* __restrict__ out, int C) {
  int i = blockIdx.x * blockDim.x + threadIdx.x;   // over B*C
  if (i >= BATCH * C) return;
  int b = i / C, d = i - b * C;
  const h16* f = feat + ((size_t)b * 128) * C + d;
  float mx = -3.0e38f;
  for (int k = 0; k < 128; ++k) mx = fmaxf(mx, (float)f[(size_t)k * C]);
  out[i] = mx;
}

// ======================= host orchestration =======================
struct LayerP { const float *W, *b, *g, *bt; int cin, cout; };

static void run_mlp(const LayerP* Ls, int nl, int kpad0, int M,
                    h16* actA, h16* actB, h16* Wt, float* tb,
                    hipStream_t stream, h16** feat, int* cfinal) {
  h16* cur = actA; h16* nxt = actB;
  int kp = kpad0;
  for (int l = 0; l < nl; ++l) {
    int cout = Ls[l].cout;
    int tot = kp * cout;
    k_prep_w<<<(tot + 255) / 256, 256, 0, stream>>>(Ls[l].W, Ls[l].b, Ls[l].g, Ls[l].bt,
                                                    Wt, tb, Ls[l].cin, kp, cout);
    dim3 grid(M / 128, cout / 32);
    k_gemm_relu<<<grid, 256, 0, stream>>>(cur, Wt, tb, nxt, kp, cout);
    h16* t = cur; cur = nxt; nxt = t;
    kp = cout;
  }
  *feat = cur;
  *cfinal = kp;
}

extern "C" void kernel_launch(void* const* d_in, const int* in_sizes, int n_in,
                              void* d_out, int out_size, void* d_ws, size_t ws_size,
                              hipStream_t stream) {
  (void)in_sizes; (void)n_in; (void)out_size; (void)ws_size;
  const float* x = (const float*)d_in[0];

  // ---- walk parameter arrays in setup_inputs() order: W, b, g, bt per layer ----
  int pi = 1;
  LayerP sa1[3][3], sa2[3][3], sa3[3];
  const int sa1d[3][4] = {{3, 32, 32, 64}, {3, 64, 64, 128}, {3, 64, 96, 128}};
  const int sa2d[3][4] = {{323, 64, 64, 128}, {323, 128, 128, 256}, {323, 128, 128, 256}};
  const int sa3d[4] = {643, 256, 512, 1024};
  for (int sc = 0; sc < 3; ++sc)
    for (int l = 0; l < 3; ++l) {
      sa1[sc][l].W = (const float*)d_in[pi++]; sa1[sc][l].b  = (const float*)d_in[pi++];
      sa1[sc][l].g = (const float*)d_in[pi++]; sa1[sc][l].bt = (const float*)d_in[pi++];
      sa1[sc][l].cin = sa1d[sc][l]; sa1[sc][l].cout = sa1d[sc][l + 1];
    }
  for (int sc = 0; sc < 3; ++sc)
    for (int l = 0; l < 3; ++l) {
      sa2[sc][l].W = (const float*)d_in[pi++]; sa2[sc][l].b  = (const float*)d_in[pi++];
      sa2[sc][l].g = (const float*)d_in[pi++]; sa2[sc][l].bt = (const float*)d_in[pi++];
      sa2[sc][l].cin = sa2d[sc][l]; sa2[sc][l].cout = sa2d[sc][l + 1];
    }
  for (int l = 0; l < 3; ++l) {
    sa3[l].W = (const float*)d_in[pi++]; sa3[l].b  = (const float*)d_in[pi++];
    sa3[l].g = (const float*)d_in[pi++]; sa3[l].bt = (const float*)d_in[pi++];
    sa3[l].cin = sa3d[l]; sa3[l].cout = sa3d[l + 1];
  }

  // ---- workspace carve ----
  uint8_t* wsp = (uint8_t*)d_ws;
  size_t off = 0;
  auto carve = [&](size_t bytes) -> void* {
    void* p = wsp + off;
    off += (bytes + 255) & ~(size_t)255;
    return p;
  };
  float* xyz   = (float*)carve((size_t)BATCH * NPTS * 3 * 4);
  int*   fidx  = (int*)carve((size_t)BATCH * 512 * 4);
  float* nxyz1 = (float*)carve((size_t)BATCH * 512 * 3 * 4);
  float* nxyz2 = (float*)carve((size_t)BATCH * 128 * 3 * 4);
  int*   gi    = (int*)carve((size_t)BATCH * 512 * 128 * 4);
  h16*   l1    = (h16*)carve((size_t)BATCH * 512 * 320 * 2);
  h16*   l2    = (h16*)carve((size_t)BATCH * 128 * 640 * 2);
  h16*   Wt    = (h16*)carve((size_t)524288 * 2);
  float* tb    = (float*)carve((size_t)1024 * 4);
  h16*   actA  = (h16*)carve((size_t)33554432 * 2);   // 64 MB
  h16*   actB  = (h16*)carve((size_t)33554432 * 2);   // 64 MB

  // ---- input layout transform ----
  k_transpose<<<(BATCH * NPTS + 255) / 256, 256, 0, stream>>>(x, xyz, NPTS);

  // ================= SA1 (xyz: B x 8192, S=512) =================
  const int S1 = 512;
  k_fps<<<BATCH, 256, 0, stream>>>(xyz, fidx, NPTS, S1);
  k_gather_xyz<<<(BATCH * S1 + 255) / 256, 256, 0, stream>>>(xyz, fidx, nxyz1, NPTS, S1);
  {
    const float rad[3] = {0.1f, 0.2f, 0.4f};
    const int   Kv[3]  = {16, 32, 128};
    const int   col[3] = {0, 64, 192};
    const int ROWS = 2048;                            // chunk of (b,s) rows
    for (int sc = 0; sc < 3; ++sc) {
      int K = Kv[sc];
      k_query_ball<<<(BATCH * S1 + 255) / 256, 256, 0, stream>>>(
          xyz, nxyz1, gi, BATCH * S1, S1, NPTS, K, rad[sc] * rad[sc]);
      for (int row0 = 0; row0 < BATCH * S1; row0 += ROWS) {
        int M = ROWS * K;
        k_group_sa1<<<(M + 255) / 256, 256, 0, stream>>>(xyz, nxyz1, gi, actA,
                                                         row0, ROWS, K, S1, NPTS);
        h16* feat; int C;
        run_mlp(sa1[sc], 3, 32, M, actA, actB, Wt, tb, stream, &feat, &C);
        k_maxpool_cat<<<(ROWS * C + 255) / 256, 256, 0, stream>>>(
            feat, l1, row0, ROWS, K, C, 320, col[sc]);
      }
    }
  }

  // ================= SA2 (xyz: B x 512, points D=320, S=128) =================
  const int S2 = 128;
  k_fps<<<BATCH, 256, 0, stream>>>(nxyz1, fidx, S1, S2);
  k_gather_xyz<<<(BATCH * S2 + 255) / 256, 256, 0, stream>>>(nxyz1, fidx, nxyz2, S1, S2);
  {
    const float rad[3] = {0.2f, 0.4f, 0.8f};
    const int   Kv[3]  = {32, 64, 128};
    const int   col[3] = {0, 128, 384};
    const int ROWS = 512;
    for (int sc = 0; sc < 3; ++sc) {
      int K = Kv[sc];
      k_query_ball<<<(BATCH * S2 + 255) / 256, 256, 0, stream>>>(
          nxyz1, nxyz2, gi, BATCH * S2, S2, S1, K, rad[sc] * rad[sc]);
      for (int row0 = 0; row0 < BATCH * S2; row0 += ROWS) {
        int M = ROWS * K;
        k_group_sa2<<<(M + 255) / 256, 256, 0, stream>>>(
            nxyz1, nxyz2, l1, gi, actA, row0, ROWS, K, S2, S1, 320, 352);
        h16* feat; int C;
        run_mlp(sa2[sc], 3, 352, M, actA, actB, Wt, tb, stream, &feat, &C);
        k_maxpool_cat<<<(ROWS * C + 255) / 256, 256, 0, stream>>>(
            feat, l2, row0, ROWS, K, C, 640, col[sc]);
      }
    }
  }

  // ================= SA3 (group_all: [B,128,643] -> MLP -> max) =================
  {
    int M = BATCH * 128;   // 2048
    k_group_sa3<<<(M + 255) / 256, 256, 0, stream>>>(nxyz2, l2, actA, M, 640, 672);
    h16* feat; int C;
    run_mlp(sa3, 3, 672, M, actA, actB, Wt, tb, stream, &feat, &C);
    k_maxpool_final<<<(BATCH * 1024 + 255) / 256, 256, 0, stream>>>(feat, (float*)d_out, 1024);
  }
}
